// CompressedSparseAttention_8615704396091
// MI455X (gfx1250) — compile-verified, matrix operand order checked by reading, not run on hardware
//
#include <hip/hip_runtime.h>
#include <hip/hip_bf16.h>
#include <math.h>

// ---------------------------------------------------------------------------
// Compressed sparse attention for MI455X (gfx1250, wave32).
// All matrix math via v_wmma_f32_16x16x32_bf16. Weights are pre-swizzled into
// fragment-major bf16 so every WMMA operand is a single contiguous 32-byte
// load; activations chain between GEMMs in bf16.
// ---------------------------------------------------------------------------

typedef __attribute__((ext_vector_type(16))) __bf16 v16bf;
typedef __attribute__((ext_vector_type(8)))  float  v8f;

#define B_    2
#define T_    4096
#define D_    1024
#define P_    1024
#define LMAX_ 16
#define H_    16
#define C_    64
#define DC_   256
#define TOPK_ 32
#define NIH_  4
#define CI_   64

__device__ __forceinline__ v8f wmma_bf16(v16bf a, v16bf b, v8f c) {
  return __builtin_amdgcn_wmma_f32_16x16x32_bf16(false, a, false, b,
                                                 (short)0, c, false, false);
}

// ---------------------------------------------------------------------------
// One-time converts / swizzles
// ---------------------------------------------------------------------------
__global__ void k_cvt(const float* __restrict__ src, __bf16* __restrict__ dst, int n) {
  const int i = blockIdx.x * blockDim.x + threadIdx.x;
  if (i < n) dst[i] = (__bf16)src[i];
}

// W (K x N, f32 row-major) -> fragment-major bf16:
// out[(((ks*ntiles + nt)*32 + lane)*16 + e] = W[ks*32 + (lane>>4)*16 + e][nt*16 + (lane&15)]
__global__ void k_swz(const float* __restrict__ W, int K, int N, __bf16* __restrict__ out) {
  const size_t i = (size_t)blockIdx.x * blockDim.x + threadIdx.x;
  if (i >= (size_t)K * N) return;
  const int e    = (int)(i & 15);
  const int lane = (int)((i >> 4) & 31);
  const size_t f = i >> 9;
  const int ntiles = N >> 4;
  const int ks = (int)(f / ntiles), nt = (int)(f % ntiles);
  const int k = ks * 32 + (lane >> 4) * 16 + e;
  const int n = nt * 16 + (lane & 15);
  out[i] = (__bf16)W[(size_t)k * N + n];
}

// ---------------------------------------------------------------------------
// Single-wave GEMM, 32x64 tile: Out = A[32 x K] * B[K x 64].
// A: bf16 row-major. Bsw: fragment-major bf16 (full-width weight, ntiles).
// OutF (f32) and/or OutB (bf16) may be null.
// ---------------------------------------------------------------------------
__global__ __launch_bounds__(32)
void k_gemm(const __bf16* __restrict__ A, int lda,
            const __bf16* __restrict__ Bsw, int ntiles, int K,
            float* __restrict__ OutF, __bf16* __restrict__ OutB, int ldo) {
  const int mt = blockIdx.x, ng = blockIdx.y;
  const int lane = threadIdx.x & 31;
  const int ln16 = lane & 15, kh = lane >> 4;
  const int ntbase = ng * 4;
  A += (size_t)mt * 32 * lda;
  const size_t obase = (size_t)mt * 32 * ldo + (size_t)ng * 64;

  v8f acc0[4] = {}, acc1[4] = {};
  const __bf16* a0p = A + (size_t)ln16 * lda + kh * 16;
  const __bf16* a1p = A + (size_t)(16 + ln16) * lda + kh * 16;
  for (int ks = 0; ks < (K >> 5); ++ks) {
    const v16bf a0 = *(const v16bf*)(a0p + (size_t)ks * 32);
    const v16bf a1 = *(const v16bf*)(a1p + (size_t)ks * 32);
#pragma unroll
    for (int nt = 0; nt < 4; ++nt) {
      const v16bf b = *(const v16bf*)(Bsw +
          ((((size_t)ks * ntiles) + ntbase + nt) * 32 + lane) * 16);
      acc0[nt] = wmma_bf16(a0, b, acc0[nt]);
      acc1[nt] = wmma_bf16(a1, b, acc1[nt]);
    }
  }
#pragma unroll
  for (int nt = 0; nt < 4; ++nt)
#pragma unroll
    for (int r = 0; r < 8; ++r) {
      const int col = nt * 16 + ln16;
      const size_t o0 = obase + (size_t)(kh * 8 + r) * ldo + col;
      const size_t o1 = obase + (size_t)(16 + kh * 8 + r) * ldo + col;
      if (OutF) { OutF[o0] = acc0[nt][r]; OutF[o1] = acc1[nt][r]; }
      if (OutB) { OutB[o0] = (__bf16)acc0[nt][r]; OutB[o1] = (__bf16)acc1[nt][r]; }
    }
}

// ---------------------------------------------------------------------------
// Phrase compression. One wave/phrase; A fragments gathered straight from the
// bf16 h copy (contiguous 32B), B fragments from swizzled W_kv / W_z.
// ---------------------------------------------------------------------------
__global__ __launch_bounds__(32)
void k_compress(const __bf16* __restrict__ hb,
                const unsigned char* __restrict__ mask,
                const int* __restrict__ tokidx,
                const __bf16* __restrict__ WkvS,
                const __bf16* __restrict__ WzS,
                const float* __restrict__ Bpos,
                float* __restrict__ outF,      // (B,P,64) or null
                __bf16* __restrict__ outB) {   // (B,P,64) or null
  __shared__ int s_tok[LMAX_];
  __shared__ int s_msk[LMAX_];
  const int p = blockIdx.x, b = blockIdx.y;
  const int lane = threadIdx.x & 31;
  if (lane < LMAX_) {
    s_tok[lane] = tokidx[((size_t)b * P_ + p) * LMAX_ + lane];
    s_msk[lane] = mask[((size_t)b * P_ + p) * LMAX_ + lane] ? 1 : 0;
  }
  __syncthreads();
  const int ln16 = lane & 15, kh = lane >> 4;
  const __bf16* arow = hb + ((size_t)b * T_ + s_tok[ln16]) * D_ + kh * 16;

  v8f ct[4] = {}, zt[4] = {};
  for (int ks = 0; ks < (D_ >> 5); ++ks) {
    const v16bf a = *(const v16bf*)(arow + (size_t)ks * 32);
#pragma unroll
    for (int nt = 0; nt < 4; ++nt) {
      const size_t fo = ((((size_t)ks * 4) + nt) * 32 + lane) * 16;
      const v16bf bw = *(const v16bf*)(WkvS + fo);
      const v16bf bz = *(const v16bf*)(WzS + fo);
      ct[nt] = wmma_bf16(a, bw, ct[nt]);
      zt[nt] = wmma_bf16(a, bz, zt[nt]);
    }
  }

  int anym = 0;
#pragma unroll
  for (int l = 0; l < LMAX_; ++l) anym |= s_msk[l];
  const int rbase = kh * 8;
#pragma unroll
  for (int nt = 0; nt < 4; ++nt) {
    const int col = nt * 16 + ln16;
    float zv[8];
    float mx = -INFINITY;
#pragma unroll
    for (int r = 0; r < 8; ++r) {
      zv[r] = zt[nt][r] + Bpos[(size_t)(rbase + r) * C_ + col];
      if (s_msk[rbase + r]) mx = fmaxf(mx, zv[r]);
    }
    mx = fmaxf(mx, __shfl_xor(mx, 16, 32));
    float se = 0.f, ac = 0.f;
#pragma unroll
    for (int r = 0; r < 8; ++r) {
      if (s_msk[rbase + r]) {
        const float e = __expf(zv[r] - mx);
        se += e; ac += e * ct[nt][r];
      }
    }
    se += __shfl_xor(se, 16, 32);
    ac += __shfl_xor(ac, 16, 32);
    const float o = (anym && se > 0.f) ? ac / se : 0.f;
    if (kh == 0) {
      const size_t oi = ((size_t)b * P_ + p) * C_ + col;
      if (outF) outF[oi] = o;
      if (outB) outB[oi] = (__bf16)o;
    }
  }
}

// ---------------------------------------------------------------------------
// wgt = h @ W_w  (N = 4 -> scalar dots)
// ---------------------------------------------------------------------------
__global__ void k_hw(const float* __restrict__ h, const float* __restrict__ Ww,
                     float* __restrict__ wgt) {
  const int i = blockIdx.x * blockDim.x + threadIdx.x;
  if (i >= B_ * T_ * NIH_) return;
  const int bt = i >> 2, hh = i & 3;
  const float* row = h + (size_t)bt * D_;
  float s = 0.f;
  for (int k = 0; k < D_; ++k) s += row[k] * Ww[(size_t)k * NIH_ + hh];
  wgt[i] = s;
}

// ---------------------------------------------------------------------------
// scores[b,t,p] = sum_hh relu(q_i·k_idx) * wgt, causal-masked. A = qi_bf
// (B*T*4 x 64), B = k_idx_bf rows (contiguous fragments).
// ---------------------------------------------------------------------------
__global__ __launch_bounds__(32)
void k_scores(const __bf16* __restrict__ qiB,
              const __bf16* __restrict__ kidxB,
              const float* __restrict__ wgt,
              const int* __restrict__ endpos,
              float* __restrict__ scores) {
  const int mt = blockIdx.x;     // 1024 tiles of 16 rows per batch
  const int pg = blockIdx.y;     // 16 groups of 64 phrases
  const int b  = blockIdx.z;
  const int lane = threadIdx.x & 31, ln16 = lane & 15, kh = lane >> 4;
  const __bf16* A = qiB + ((size_t)b * T_ * NIH_ + (size_t)mt * 16) * CI_;
  v8f acc[4] = {};
#pragma unroll
  for (int ks = 0; ks < 2; ++ks) {
    const int k = ks * 32 + kh * 16;
    const v16bf a = *(const v16bf*)(A + (size_t)ln16 * CI_ + k);
#pragma unroll
    for (int nt = 0; nt < 4; ++nt) {
      const int p = pg * 64 + nt * 16 + ln16;
      const v16bf bfr = *(const v16bf*)(kidxB + ((size_t)b * P_ + p) * CI_ + k);
      acc[nt] = wmma_bf16(a, bfr, acc[nt]);
    }
  }
  const int t0 = mt * 4 + kh * 2;
#pragma unroll
  for (int nt = 0; nt < 4; ++nt) {
    const int p = pg * 64 + nt * 16 + ln16;
    const int end = endpos[(size_t)b * P_ + p];
    float s0 = 0.f, s1 = 0.f;
#pragma unroll
    for (int r = 0; r < 4; ++r)
      s0 += fmaxf(acc[nt][r], 0.f) * wgt[((size_t)b * T_ + t0) * NIH_ + r];
#pragma unroll
    for (int r = 4; r < 8; ++r)
      s1 += fmaxf(acc[nt][r], 0.f) * wgt[((size_t)b * T_ + t0 + 1) * NIH_ + (r - 4)];
    scores[((size_t)b * T_ + t0) * P_ + p]     = (end < t0)     ? s0 : -INFINITY;
    scores[((size_t)b * T_ + t0 + 1) * P_ + p] = (end < t0 + 1) ? s1 : -INFINITY;
  }
}

// ---------------------------------------------------------------------------
// Top-32 over P=1024 per (b,t): 32 wave-wide argmax rounds.
// ---------------------------------------------------------------------------
__global__ __launch_bounds__(32)
void k_topk(const float* __restrict__ scores, int* __restrict__ tidx,
            float* __restrict__ tsc) {
  const int bt = blockIdx.x;
  const int lane = threadIdx.x & 31;
  const float* row = scores + (size_t)bt * P_;
  float v[32];
#pragma unroll
  for (int j = 0; j < 32; ++j) v[j] = row[lane * 32 + j];
  for (int sel = 0; sel < TOPK_; ++sel) {
    float lm = -INFINITY; int li = 0;
#pragma unroll
    for (int j = 0; j < 32; ++j) if (v[j] > lm) { lm = v[j]; li = j; }
    int gi = lane * 32 + li;
#pragma unroll
    for (int off = 16; off > 0; off >>= 1) {
      const float om = __shfl_xor(lm, off, 32);
      const int   oi = __shfl_xor(gi, off, 32);
      if (om > lm || (om == lm && oi < gi)) { lm = om; gi = oi; }
    }
    if (lane == 0) { tidx[(size_t)bt * TOPK_ + sel] = gi; tsc[(size_t)bt * TOPK_ + sel] = lm; }
    if ((gi >> 5) == lane) {
      const int kj = gi & 31;
#pragma unroll
      for (int j = 0; j < 32; ++j) if (j == kj) v[j] = -INFINITY;
    }
  }
}

// ---------------------------------------------------------------------------
// Per-token attention. One wave per (b,t); logits + attn@k via WMMA; output
// written directly as bf16 (A operand of the final W_o GEMM).
// ---------------------------------------------------------------------------
__global__ __launch_bounds__(32)
void k_attn(const float* __restrict__ qh,      // (B*T, 1024) f32
            const float* __restrict__ ccomp,   // (B,P,64)  f32
            const int* __restrict__ endpos,
            const int* __restrict__ tidx, const float* __restrict__ tsc,
            const float* __restrict__ cosT, const float* __restrict__ sinT,
            const float* __restrict__ qnw, const float* __restrict__ knw,
            const float* __restrict__ sink,
            __bf16* __restrict__ outB /* (B*T, 1024) bf16 */) {
  __shared__ __align__(32) __bf16 q_lds[H_][C_];
  __shared__ __align__(32) __bf16 k_lds[TOPK_][C_];
  __shared__ __align__(32) __bf16 a_lds[H_][TOPK_];
  __shared__ float qscale[H_];
  __shared__ float kvalid[TOPK_];
  const int bt = blockIdx.x;
  const int b = bt / T_, t = bt % T_;
  const int lane = threadIdx.x & 31;
  const float* qrow = qh + (size_t)bt * (H_ * C_);

  if (lane < H_) {
    float ss = 0.f;
    for (int c = 0; c < C_; ++c) { const float x = qrow[lane * C_ + c]; ss += x * x; }
    qscale[lane] = rsqrtf(ss / (float)C_ + 1e-6f);
  }
  {
    const int kk = lane;
    const int pidx = tidx[(size_t)bt * TOPK_ + kk];
    const float sc = tsc[(size_t)bt * TOPK_ + kk];
    kvalid[kk] = (sc >= -1e30f) ? 1.f : 0.f;
    const int ep = endpos[(size_t)b * P_ + pidx];
    const int pos = ep > 0 ? ep : 0;
    const float* kr = ccomp + ((size_t)b * P_ + pidx) * C_;
    float ss = 0.f;
    for (int c = 0; c < C_; ++c) { const float x = kr[c]; ss += x * x; }
    const float scl = rsqrtf(ss / (float)C_ + 1e-6f);
    const float* cr = cosT + (size_t)pos * C_;
    const float* sr = sinT + (size_t)pos * C_;
    for (int c = 0; c < C_; ++c) {
      const int cp = (c < 32) ? c + 32 : c - 32;
      const float xn = kr[c]  * scl * knw[c];
      const float xo = kr[cp] * scl * knw[cp];
      k_lds[kk][c] = (__bf16)(xn * cr[c] + ((c < 32) ? -xo : xo) * sr[c]);
    }
  }
  __syncthreads();
  {
    const float* cr = cosT + (size_t)t * C_;
    const float* sr = sinT + (size_t)t * C_;
    for (int i = lane; i < H_ * C_; i += 32) {
      const int hh = i >> 6, c = i & 63;
      const int cp = (c < 32) ? c + 32 : c - 32;
      const float s = qscale[hh];
      const float xn = qrow[i] * s * qnw[c];
      const float xo = qrow[hh * C_ + cp] * s * qnw[cp];
      q_lds[hh][c] = (__bf16)(xn * cr[c] + ((c < 32) ? -xo : xo) * sr[c]);
    }
  }
  __syncthreads();

  const int ln16 = lane & 15, kh = lane >> 4;
  v8f lg[2] = {};
#pragma unroll
  for (int ks = 0; ks < 2; ++ks) {
    const v16bf a = *(const v16bf*)&q_lds[ln16][ks * 32 + kh * 16];
#pragma unroll
    for (int nt = 0; nt < 2; ++nt) {
      const v16bf bfr = *(const v16bf*)&k_lds[nt * 16 + ln16][ks * 32 + kh * 16];
      lg[nt] = wmma_bf16(a, bfr, lg[nt]);
    }
  }
#pragma unroll
  for (int nt = 0; nt < 2; ++nt) {
    const float kv = kvalid[nt * 16 + ln16];
#pragma unroll
    for (int r = 0; r < 8; ++r) {
      const float l = lg[nt][r] * 0.125f;
      lg[nt][r] = (kv > 0.f) ? l : -INFINITY;
    }
  }
#pragma unroll
  for (int r = 0; r < 8; ++r) {
    const int hh = kh * 8 + r;
    float m = fmaxf(lg[0][r], lg[1][r]);
#pragma unroll
    for (int off = 1; off < 16; off <<= 1) m = fmaxf(m, __shfl_xor(m, off, 32));
    const float sk = sink[hh];
    m = fmaxf(m, sk);
    const float e0 = __expf(lg[0][r] - m);
    const float e1 = __expf(lg[1][r] - m);
    float s = e0 + e1;
#pragma unroll
    for (int off = 1; off < 16; off <<= 1) s += __shfl_xor(s, off, 32);
    s += __expf(sk - m);
    const float inv = 1.f / s;
    a_lds[hh][ln16]      = (__bf16)(e0 * inv);
    a_lds[hh][16 + ln16] = (__bf16)(e1 * inv);
  }
  __syncthreads();
  const v16bf a2 = *(const v16bf*)&a_lds[ln16][kh * 16];
  v8f oc[4] = {};
#pragma unroll
  for (int nt = 0; nt < 4; ++nt) {
    v16bf bfr;
#pragma unroll
    for (int j = 0; j < 16; ++j) bfr[j] = k_lds[kh * 16 + j][nt * 16 + ln16];
    oc[nt] = wmma_bf16(a2, bfr, oc[nt]);
  }
#pragma unroll
  for (int nt = 0; nt < 4; ++nt)
#pragma unroll
    for (int r = 0; r < 8; ++r) {
      const int hh = kh * 8 + r;
      outB[(size_t)bt * (H_ * C_) + hh * C_ + nt * 16 + ln16] = (__bf16)oc[nt][r];
    }
}

// ---------------------------------------------------------------------------
extern "C" void kernel_launch(void* const* d_in, const int* in_sizes, int n_in,
                              void* d_out, int out_size, void* d_ws, size_t ws_size,
                              hipStream_t stream) {
  (void)in_sizes; (void)n_in; (void)out_size; (void)ws_size;
  const float* h        = (const float*)d_in[0];
  const unsigned char* pmask = (const unsigned char*)d_in[1];
  const int*   ptok     = (const int*)d_in[2];
  const int*   pend     = (const int*)d_in[3];
  const float* rcos     = (const float*)d_in[4];
  const float* rsin     = (const float*)d_in[5];
  const float* W_dq     = (const float*)d_in[6];
  const float* W_uq     = (const float*)d_in[7];
  const float* W_kv     = (const float*)d_in[8];
  const float* W_z      = (const float*)d_in[9];
  const float* B_pos    = (const float*)d_in[10];
  const float* W_kvi    = (const float*)d_in[11];
  const float* W_zi     = (const float*)d_in[12];
  const float* B_posi   = (const float*)d_in[13];
  const float* W_iuq    = (const float*)d_in[14];
  const float* W_w      = (const float*)d_in[15];
  const float* q_norm_w = (const float*)d_in[16];
  const float* k_norm_w = (const float*)d_in[17];
  const float* W_o      = (const float*)d_in[18];
  const float* sinkl    = (const float*)d_in[19];
  float* out = (float*)d_out;

  // bump allocator over d_ws (256B aligned chunks)
  char* wp = (char*)d_ws;
  auto alloc = [&wp](size_t bytes) -> char* {
    char* p = wp; wp += (bytes + 255) & ~(size_t)255; return p;
  };
  const size_t NT = (size_t)B_ * T_;                        // 8192
  __bf16* h_bf    = (__bf16*)alloc(NT * D_ * 2);            // 16 MB
  __bf16* qlat_bf = (__bf16*)alloc(NT * DC_ * 2);           // 4 MB
  __bf16* qi_bf   = (__bf16*)alloc(NT * NIH_ * CI_ * 2);    // 4 MB
  float*  qh_f    = (float*)alloc(NT * H_ * C_ * 4);        // 32 MB
  float*  wgt_f   = (float*)alloc(NT * NIH_ * 4);
  float*  ccomp_f = (float*)alloc((size_t)B_ * P_ * C_ * 4);
  __bf16* kidx_bf = (__bf16*)alloc((size_t)B_ * P_ * CI_ * 2);
  __bf16* WdqS    = (__bf16*)alloc((size_t)D_ * DC_ * 2);
  __bf16* WuqS    = (__bf16*)alloc((size_t)DC_ * H_ * C_ * 2);
  __bf16* WiuqS   = (__bf16*)alloc((size_t)DC_ * NIH_ * CI_ * 2);
  __bf16* WoS     = (__bf16*)alloc((size_t)H_ * C_ * D_ * 2);
  __bf16* WkvS    = (__bf16*)alloc((size_t)D_ * C_ * 2);
  __bf16* WzS     = (__bf16*)alloc((size_t)D_ * C_ * 2);
  __bf16* WkviS   = (__bf16*)alloc((size_t)D_ * CI_ * 2);
  __bf16* WziS    = (__bf16*)alloc((size_t)D_ * CI_ * 2);
  float*  scores  = (float*)alloc(NT * P_ * 4);             // 32 MB
  float*  tsc     = (float*)alloc(NT * TOPK_ * 4);
  int*    tidx    = (int*)alloc(NT * TOPK_ * 4);
  __bf16* attn_bf = (__bf16*)scores;   // alias: scores dead after top-k

  const dim3 wave(32);
  const int thr = 256;

  // 0) one-time converts / weight swizzles
  k_cvt<<<dim3((NT * D_ + thr - 1) / thr), dim3(thr), 0, stream>>>(h, h_bf, (int)(NT * D_));
  k_swz<<<dim3((D_ * DC_ + thr - 1) / thr), dim3(thr), 0, stream>>>(W_dq, D_, DC_, WdqS);
  k_swz<<<dim3((DC_ * H_ * C_ + thr - 1) / thr), dim3(thr), 0, stream>>>(W_uq, DC_, H_ * C_, WuqS);
  k_swz<<<dim3((DC_ * NIH_ * CI_ + thr - 1) / thr), dim3(thr), 0, stream>>>(W_iuq, DC_, NIH_ * CI_, WiuqS);
  k_swz<<<dim3((H_ * C_ * D_ + thr - 1) / thr), dim3(thr), 0, stream>>>(W_o, H_ * C_, D_, WoS);
  k_swz<<<dim3((D_ * C_ + thr - 1) / thr), dim3(thr), 0, stream>>>(W_kv, D_, C_, WkvS);
  k_swz<<<dim3((D_ * C_ + thr - 1) / thr), dim3(thr), 0, stream>>>(W_z, D_, C_, WzS);
  k_swz<<<dim3((D_ * CI_ + thr - 1) / thr), dim3(thr), 0, stream>>>(W_kvi, D_, CI_, WkviS);
  k_swz<<<dim3((D_ * CI_ + thr - 1) / thr), dim3(thr), 0, stream>>>(W_zi, D_, CI_, WziS);

  // 1) phrase compression: c_comp (f32) and k_idx (bf16)
  k_compress<<<dim3(P_, B_), wave, 0, stream>>>(h_bf, pmask, ptok, WkvS, WzS, B_pos,
                                                ccomp_f, (__bf16*)nullptr);
  k_compress<<<dim3(P_, B_), wave, 0, stream>>>(h_bf, pmask, ptok, WkviS, WziS, B_posi,
                                                (float*)nullptr, kidx_bf);

  // 2) q_latent(bf16) = h @ W_dq
  k_gemm<<<dim3(256, 4), wave, 0, stream>>>(h_bf, D_, WdqS, DC_ / 16, D_,
                                            (float*)nullptr, qlat_bf, DC_);
  // 3) qh(f32) = q_latent @ W_uq
  k_gemm<<<dim3(256, 16), wave, 0, stream>>>(qlat_bf, DC_, WuqS, (H_ * C_) / 16, DC_,
                                             qh_f, (__bf16*)nullptr, H_ * C_);
  // 4) qi(bf16) = q_latent @ W_iuq
  k_gemm<<<dim3(256, 4), wave, 0, stream>>>(qlat_bf, DC_, WiuqS, (NIH_ * CI_) / 16, DC_,
                                            (float*)nullptr, qi_bf, NIH_ * CI_);
  // 5) wgt = h @ W_w
  k_hw<<<dim3((B_ * T_ * NIH_ + thr - 1) / thr), dim3(thr), 0, stream>>>(h, W_w, wgt_f);

  // 6) phrase-selection scores (causal masked)
  k_scores<<<dim3(T_ * NIH_ / 16, P_ / 64, B_), wave, 0, stream>>>(qi_bf, kidx_bf, wgt_f,
                                                                   pend, scores);
  // 7) top-32 per token
  k_topk<<<dim3((int)NT), wave, 0, stream>>>(scores, tidx, tsc);

  // 8) per-token attention (writes bf16)
  k_attn<<<dim3((int)NT), wave, 0, stream>>>(qh_f, ccomp_f, pend, tidx, tsc, rcos, rsin,
                                             q_norm_w, k_norm_w, sinkl, attn_bf);

  // 9) out(f32) = attnout @ W_o
  k_gemm<<<dim3(256, 16), wave, 0, stream>>>(attn_bf, H_ * C_, WoS, D_ / 16, H_ * C_,
                                             out, (__bf16*)nullptr, D_);
}
